// PathNodeEncoder_46239617909499
// MI455X (gfx1250) — matrix operand order repaired and checked by reading, not compiled, not run on hardware
//
#include <hip/hip_runtime.h>
#include <hip/hip_bf16.h>

// ---------------------------------------------------------------------------
// PathNodeEncoder forward on gfx1250 (MI455X).
// Big GEMMs -> v_wmma_f32_16x16x32_f16, 128x64 block tile, 32x32 per wave
//              (4 WMMA / 8 ds_load_b128 per wave per K-step),
//              register-prefetched, vectorized b128 global loads.
// Attention  -> online-softmax, wave-per-query, shuffle reductions.
// ---------------------------------------------------------------------------

typedef __attribute__((ext_vector_type(16))) _Float16 v16h;
typedef __attribute__((ext_vector_type(8)))  float    v8f;
typedef __attribute__((ext_vector_type(4)))  float    f32x4;
typedef __attribute__((ext_vector_type(4)))  _Float16 f16x4;

// ----- static problem sizes (match reference) -----
#define C_B      16
#define C_U      128
#define C_P      32
#define C_TTOK   24
#define C_TPATH  48
#define C_D      512
#define C_FFN    2048
#define C_HEADS  8
#define C_NU     2048        // B*U
#define C_NP     512         // B*P
#define C_4H     2048
#define C_OUTEL  (C_U * C_B * C_D)   // 1048576

// ===========================================================================
// WMMA GEMM: C[M,N] = A[M,K] @ B[K,N]  (+bias[n]) (+=C if accum) (relu opt)
// transB: B[k][n] = Bm[n*K + k]  (for  x @ W^T  with row-major W[N,K])
// Requires M%128==0, N%64==0, K%32==0, all pointers 16B aligned.
// Block tile 128x64, 8 waves, each wave a 32x32 sub-tile (4 WMMAs).
// ===========================================================================
__global__ void __launch_bounds__(256)
k_gemm(const float* __restrict__ A, const float* __restrict__ Bm,
       const float* __restrict__ bias, float* __restrict__ C,
       int M, int N, int K, int transB, int relu, int accum)
{
  __shared__ __align__(16) _Float16 As[128][40];  // [m][k], row stride 80B
  __shared__ __align__(16) _Float16 Bs[64][40];   // [n][k], row stride 80B

  const int tid  = threadIdx.x;
  const int lane = tid & 31;
  const int wv   = tid >> 5;          // 0..7
  const int m0   = blockIdx.y * 128;
  const int n0   = blockIdx.x * 64;
  const int wm   = wv & 3;            // 32-row block within 128
  const int wn   = wv >> 2;           // 32-col block within 64

  // staging coordinates
  const int ar  = tid >> 3;           // 0..31  (A rows ar, +32, +64, +96)
  const int ac4 = tid & 7;            // float4 index within 32-wide K slab
  const int bkk = tid >> 4;           // non-trans B: k row 0..15 (+16)
  const int bn4 = (tid & 15) * 4;     // non-trans B: n column group

  f32x4 ra0, ra1, ra2, ra3, rb0, rb1;

#define LOAD_TILES(K0)                                                           \
  do {                                                                           \
    ra0 = *(const f32x4*)(A + (size_t)(m0 + ar) * K + (K0) + ac4 * 4);           \
    ra1 = *(const f32x4*)(A + (size_t)(m0 + ar + 32) * K + (K0) + ac4 * 4);      \
    ra2 = *(const f32x4*)(A + (size_t)(m0 + ar + 64) * K + (K0) + ac4 * 4);      \
    ra3 = *(const f32x4*)(A + (size_t)(m0 + ar + 96) * K + (K0) + ac4 * 4);      \
    if (transB) {                                                                \
      rb0 = *(const f32x4*)(Bm + (size_t)(n0 + ar) * K + (K0) + ac4 * 4);        \
      rb1 = *(const f32x4*)(Bm + (size_t)(n0 + ar + 32) * K + (K0) + ac4 * 4);   \
    } else {                                                                     \
      rb0 = *(const f32x4*)(Bm + (size_t)((K0) + bkk) * N + n0 + bn4);           \
      rb1 = *(const f32x4*)(Bm + (size_t)((K0) + bkk + 16) * N + n0 + bn4);      \
    }                                                                            \
  } while (0)

  v8f c00 = {}, c01 = {}, c10 = {}, c11 = {};

  LOAD_TILES(0);

  for (int k0 = 0; k0 < K; k0 += 32) {
    __syncthreads();
    // ---- store prefetched tiles to LDS (f32 -> f16, vectorized) ----
    *(f16x4*)&As[ar][ac4 * 4]      = __builtin_convertvector(ra0, f16x4);
    *(f16x4*)&As[ar + 32][ac4 * 4] = __builtin_convertvector(ra1, f16x4);
    *(f16x4*)&As[ar + 64][ac4 * 4] = __builtin_convertvector(ra2, f16x4);
    *(f16x4*)&As[ar + 96][ac4 * 4] = __builtin_convertvector(ra3, f16x4);
    if (transB) {
      *(f16x4*)&Bs[ar][ac4 * 4]      = __builtin_convertvector(rb0, f16x4);
      *(f16x4*)&Bs[ar + 32][ac4 * 4] = __builtin_convertvector(rb1, f16x4);
    } else {
#pragma unroll
      for (int j = 0; j < 4; ++j) {
        Bs[bn4 + j][bkk]      = (_Float16)rb0[j];
        Bs[bn4 + j][bkk + 16] = (_Float16)rb1[j];
      }
    }
    // ---- issue next tile's global loads early (overlap barrier+WMMA) ----
    if (k0 + 32 < K) { LOAD_TILES(k0 + 32); }
    __syncthreads();

    // ---- fragments in contiguous k order (k = 16g+j), same for A and B ----
    const int g   = lane >> 4;
    const int ml0 = 32 * wm + (lane & 15);
    const int nl0 = 32 * wn + (lane & 15);
    v16h a0, a1, b0, b1;
#pragma unroll
    for (int j = 0; j < 16; ++j) {
      a0[j]     = As[ml0][16 * g + j];           // K = 16g+j
      a1[j]     = As[ml0 + 16][16 * g + j];
    }
#pragma unroll
    for (int j = 0; j < 16; ++j) {
      b0[j] = Bs[nl0][16 * g + j];               // K = 16g+j
      b1[j] = Bs[nl0 + 16][16 * g + j];
    }

    c00 = __builtin_amdgcn_wmma_f32_16x16x32_f16(false, a0, false, b0, (short)0, c00, false, false);
    c01 = __builtin_amdgcn_wmma_f32_16x16x32_f16(false, a0, false, b1, (short)0, c01, false, false);
    c10 = __builtin_amdgcn_wmma_f32_16x16x32_f16(false, a1, false, b0, (short)0, c10, false, false);
    c11 = __builtin_amdgcn_wmma_f32_16x16x32_f16(false, a1, false, b1, (short)0, c11, false, false);
  }
#undef LOAD_TILES

  // ---- epilogue: C layout: VGPR r -> row r + 8*(lane>>4), col lane&15 ----
  const int g     = lane >> 4;
  const int mbase = m0 + 32 * wm + 8 * g;
  const int nA    = n0 + 32 * wn + (lane & 15);
  const int nB    = nA + 16;
#pragma unroll
  for (int r = 0; r < 8; ++r) {
    int mA = mbase + r;
    int mB = mA + 16;
    float v00 = c00[r], v01 = c01[r], v10 = c10[r], v11 = c11[r];
    if (bias) {
      float bA = bias[nA], bB = bias[nB];
      v00 += bA; v01 += bB; v10 += bA; v11 += bB;
    }
    if (accum) {
      v00 += C[(size_t)mA * N + nA];  v01 += C[(size_t)mA * N + nB];
      v10 += C[(size_t)mB * N + nA];  v11 += C[(size_t)mB * N + nB];
    }
    if (relu) {
      v00 = fmaxf(v00, 0.f); v01 = fmaxf(v01, 0.f);
      v10 = fmaxf(v10, 0.f); v11 = fmaxf(v11, 0.f);
    }
    C[(size_t)mA * N + nA] = v00;
    C[(size_t)mA * N + nB] = v01;
    C[(size_t)mB * N + nA] = v10;
    C[(size_t)mB * N + nB] = v11;
  }
}

// ===========================================================================
// small helpers
// ===========================================================================
__device__ __forceinline__ float posenc_val(int pos, int d)
{
  int i = d >> 1;
  float div = __expf((float)(2 * i) * (-9.210340371976184f / (float)C_D));
  float ang = (float)pos * div;
  return (d & 1) ? __cosf(ang) : __sinf(ang);
}

__global__ void k_fill(float* p, float v, int n)
{
  int i = blockIdx.x * 256 + threadIdx.x;
  if (i < n) p[i] = v;
}

__global__ void k_ulen(const int* __restrict__ units, int* __restrict__ ulen)
{
  int n = blockIdx.x * 256 + threadIdx.x;
  if (n >= C_NU) return;
  int L = C_TTOK;
  for (int t = 0; t < C_TTOK; ++t)
    if (units[t * C_NU + n] == 0) { L = t; break; }
  ulen[n] = L;
}

__global__ void k_plen(const int* __restrict__ paths, int* __restrict__ endp)
{
  int p = blockIdx.x * 256 + threadIdx.x;
  if (p >= C_NP) return;
  int L = C_TPATH;
  for (int t = 0; t < C_TPATH; ++t)
    if (paths[t * C_NP + p] == -1) { L = t; break; }
  endp[p] = L;
}

__global__ void k_embed(const int* __restrict__ units, const float* __restrict__ emb,
                        float* __restrict__ x)
{
  int idx = blockIdx.x * 256 + threadIdx.x;          // < 24*2048*512
  int d = idx & 511;
  int n = (idx >> 9) & (C_NU - 1);
  int t = idx >> 20;
  int tok = units[t * C_NU + n];
  x[idx] = emb[(size_t)tok * C_D + d];
}

__global__ void k_rev_step(float* __restrict__ xs, const float* __restrict__ x,
                           const int* __restrict__ ulen, int t)
{
  int idx = blockIdx.x * 256 + threadIdx.x;          // < 2048*512
  int n = idx >> 9;
  int d = idx & 511;
  int st = ulen[n] - 1 - t;
  if (st < 0) st = 0;
  xs[idx] = x[((size_t)st * C_NU + n) * C_D + d];
}

__global__ void k_lstm_gate(const float* __restrict__ G, float* __restrict__ h,
                            float* __restrict__ c, const int* __restrict__ ulen, int t)
{
  int idx = blockIdx.x * 256 + threadIdx.x;          // < 2048*512
  int n = idx >> 9, j = idx & 511;
  if (t >= ulen[n]) return;                          // state frozen past length
  size_t base = (size_t)n * C_4H + j;
  float gi = G[base];
  float gf = G[base + 512];
  float gg = G[base + 1024];
  float go = G[base + 1536];
  float si = 1.f / (1.f + __expf(-gi));
  float sf = 1.f / (1.f + __expf(-gf));
  float so = 1.f / (1.f + __expf(-go));
  float cn = sf * c[idx] + si * tanhf(gg);
  c[idx] = cn;
  h[idx] = so * tanhf(cn);
}

__global__ void k_hcat(const float* __restrict__ hf, const float* __restrict__ hb,
                       float* __restrict__ hc)
{
  int idx = blockIdx.x * 256 + threadIdx.x;          // < 2048*1024
  int n = idx >> 10, j = idx & 1023;
  hc[idx] = (j < 512) ? hf[(size_t)n * 512 + j] : hb[(size_t)n * 512 + (j - 512)];
}

__global__ void k_paths_meta(const int* __restrict__ paths, const int* __restrict__ endp,
                             int* __restrict__ contrib, int* __restrict__ segarr,
                             int* __restrict__ valid)
{
  int idx = blockIdx.x * 256 + threadIdx.x;          // < 48*512
  if (idx >= C_TPATH * C_NP) return;
  int t = idx / C_NP, p = idx % C_NP;
  int val = paths[idx];
  int v = (t < endp[p]) ? 1 : 0;
  valid[idx] = v;
  contrib[idx] = (v && val >= 0 && val < C_U) ? 1 : 0;
  int cv = val < 0 ? 0 : (val > C_U - 1 ? C_U - 1 : val);
  segarr[idx] = (p >> 5) * C_U + cv;
}

__global__ void k_build_ctx(const float* __restrict__ u, const int* __restrict__ contrib,
                            const int* __restrict__ segarr, float* __restrict__ ctx)
{
  int idx = blockIdx.x * 256 + threadIdx.x;          // < 48*512*512
  int d = idx & 511;
  int rest = idx >> 9;
  int p = rest & (C_NP - 1);
  int t = rest >> 9;
  int e = t * C_NP + p;
  float v = contrib[e] ? u[(size_t)segarr[e] * C_D + d] : 0.f;
  ctx[idx] = v + posenc_val(t, d);
}

__global__ void k_build_order(const float* __restrict__ u, float* __restrict__ ox)
{
  int idx = blockIdx.x * 256 + threadIdx.x;          // < 128*16*512
  int d = idx & 511;
  int rest = idx >> 9;
  int b = rest & (C_B - 1);
  int uu = rest >> 4;
  ox[idx] = u[((size_t)b * C_U + uu) * C_D + d] + posenc_val(uu, d);
}

// ---- attention: one block per (n,head); wave-per-query online softmax ----
__global__ void __launch_bounds__(128)
k_attn(const float* __restrict__ qkv, float* __restrict__ outc,
       const int* __restrict__ validk, int T, int N)
{
  const int n = blockIdx.x / C_HEADS;
  const int h = blockIdx.x % C_HEADS;
  const int wv = threadIdx.x >> 5;
  const int lane = threadIdx.x & 31;
  const int d0 = lane * 2, d1 = d0 + 1;

  for (int q = wv; q < T; q += 4) {
    const float* qp = qkv + (size_t)(q * N + n) * (3 * C_D) + h * 64;
    float q0 = qp[d0], q1 = qp[d1];
    float m = -3.0e38f, l = 0.f, a0 = 0.f, a1 = 0.f;
    for (int k = 0; k < T; ++k) {
      const float* kp = qkv + (size_t)(k * N + n) * (3 * C_D) + C_D + h * 64;
      float part = q0 * kp[d0] + q1 * kp[d1];
      for (int off = 16; off > 0; off >>= 1) part += __shfl_xor(part, off, 32);
      float s = part * 0.125f;                              // 1/sqrt(64)
      if (validk && !validk[k * N + n]) s = -1e9f;
      float mn   = fmaxf(m, s);
      float corr = __expf(m - mn);
      float w    = __expf(s - mn);
      const float* vp = qkv + (size_t)(k * N + n) * (3 * C_D) + 2 * C_D + h * 64;
      l  = l  * corr + w;
      a0 = a0 * corr + w * vp[d0];
      a1 = a1 * corr + w * vp[d1];
      m = mn;
    }
    float inv = 1.f / l;
    float* op = outc + (size_t)(q * N + n) * C_D + h * 64;
    op[d0] = a0 * inv;
    op[d1] = a1 * inv;
  }
}

// ---- x = LN(x + y) * s + b, one block per row of 512 ----
__global__ void __launch_bounds__(256)
k_ln_res(float* __restrict__ x, const float* __restrict__ y,
         const float* __restrict__ s, const float* __restrict__ b)
{
  __shared__ float red[256];
  const int row = blockIdx.x, tid = threadIdx.x;
  size_t base = (size_t)row * C_D;
  float v0 = x[base + tid]       + y[base + tid];
  float v1 = x[base + tid + 256] + y[base + tid + 256];
  red[tid] = v0 + v1; __syncthreads();
  for (int st = 128; st > 0; st >>= 1) { if (tid < st) red[tid] += red[tid + st]; __syncthreads(); }
  float mean = red[0] / (float)C_D;
  __syncthreads();
  float e0 = v0 - mean, e1 = v1 - mean;
  red[tid] = e0 * e0 + e1 * e1; __syncthreads();
  for (int st = 128; st > 0; st >>= 1) { if (tid < st) red[tid] += red[tid + st]; __syncthreads(); }
  float inv = rsqrtf(red[0] / (float)C_D + 1e-5f);
  x[base + tid]       = e0 * inv * s[tid]       + b[tid];
  x[base + tid + 256] = e1 * inv * s[tid + 256] + b[tid + 256];
}

// ---- s[r] = dot(x_row, attn_w) + attn_b ; wave per row ----
__global__ void __launch_bounds__(256)
k_score(const float* __restrict__ x, const float* __restrict__ aw,
        const float* __restrict__ ab, float* __restrict__ s, int rows, int order_layout)
{
  int wv = threadIdx.x >> 5, lane = threadIdx.x & 31;
  int r = blockIdx.x * 8 + wv;
  if (r >= rows) return;
  const float* xp;
  if (order_layout) {                       // r = b*U + uu, x is [U,B,D]
    int uu = r & (C_U - 1), bb = r >> 7;
    xp = x + ((size_t)uu * C_B + bb) * C_D;
  } else {
    xp = x + (size_t)r * C_D;
  }
  float acc = 0.f;
  for (int j = lane; j < C_D; j += 32) acc += xp[j] * aw[j];
  for (int off = 16; off > 0; off >>= 1) acc += __shfl_xor(acc, off, 32);
  if (lane == 0) s[r] = acc + ab[0];
}

// ---- masked-softmax segment merge; one block per (graph,unit); +posenc ----
__global__ void __launch_bounds__(256)
k_splice(const float* __restrict__ expaths, const float* __restrict__ order_x,
         const float* __restrict__ s_path, const float* __restrict__ s_old,
         const int* __restrict__ contrib, const int* __restrict__ segarr,
         float* __restrict__ out_x)
{
  __shared__ float red[256];
  const int segid = blockIdx.x;                 // b*U + uu
  const int b = segid >> 7, uu = segid & (C_U - 1);
  const int tid = threadIdx.x;
  const int NE = C_TPATH * C_P;                 // 1536 entries in this graph

  // max
  float lm = -3.0e38f;
  for (int e = tid; e < NE; e += 256) {
    int t = e >> 5, pl = e & 31;
    int idx = t * C_NP + b * C_P + pl;
    if (contrib[idx] && segarr[idx] == segid) lm = fmaxf(lm, s_path[idx]);
  }
  red[tid] = lm; __syncthreads();
  for (int st = 128; st > 0; st >>= 1) { if (tid < st) red[tid] = fmaxf(red[tid], red[tid + st]); __syncthreads(); }
  float m = fmaxf(red[0], s_old[segid]);
  __syncthreads();

  // denominator
  float ls = 0.f;
  for (int e = tid; e < NE; e += 256) {
    int t = e >> 5, pl = e & 31;
    int idx = t * C_NP + b * C_P + pl;
    if (contrib[idx] && segarr[idx] == segid) ls += __expf(s_path[idx] - m);
  }
  red[tid] = ls; __syncthreads();
  for (int st = 128; st > 0; st >>= 1) { if (tid < st) red[tid] += red[tid + st]; __syncthreads(); }
  float wold = __expf(s_old[segid] - m);
  float den = red[0] + wold;
  __syncthreads();

  // numerator (each thread owns dims tid and tid+256)
  size_t obase = ((size_t)uu * C_B + b) * C_D;
  float a0 = wold * order_x[obase + tid];
  float a1 = wold * order_x[obase + tid + 256];
  for (int e = 0; e < NE; ++e) {
    int t = e >> 5, pl = e & 31;
    int idx = t * C_NP + b * C_P + pl;
    if (contrib[idx] && segarr[idx] == segid) {
      float w = __expf(s_path[idx] - m);
      const float* xp = expaths + (size_t)idx * C_D;
      a0 += w * xp[tid];
      a1 += w * xp[tid + 256];
    }
  }
  float inv = 1.f / den;
  out_x[obase + tid]       = a0 * inv + posenc_val(uu, tid);
  out_x[obase + tid + 256] = a1 * inv + posenc_val(uu, tid + 256);
}

// ===========================================================================
// host-side orchestration
// ===========================================================================
struct EncParams {
  const float *qkv_w, *qkv_b, *out_w, *out_b, *ln1_s, *ln1_b,
              *ff1_w, *ff1_b, *ff2_w, *ff2_b, *ln2_s, *ln2_b;
};

static EncParams make_enc(void* const* d_in, int base)
{
  EncParams p;
  p.qkv_w = (const float*)d_in[base + 0];  p.qkv_b = (const float*)d_in[base + 1];
  p.out_w = (const float*)d_in[base + 2];  p.out_b = (const float*)d_in[base + 3];
  p.ln1_s = (const float*)d_in[base + 4];  p.ln1_b = (const float*)d_in[base + 5];
  p.ff1_w = (const float*)d_in[base + 6];  p.ff1_b = (const float*)d_in[base + 7];
  p.ff2_w = (const float*)d_in[base + 8];  p.ff2_b = (const float*)d_in[base + 9];
  p.ln2_s = (const float*)d_in[base + 10]; p.ln2_b = (const float*)d_in[base + 11];
  return p;
}

static inline int imin_(int a, int b) { return a < b ? a : b; }

static void run_encoder(float* x, int T, int N, const EncParams& p, int L,
                        const int* validk, float* qkvbuf, float* attnctx,
                        float* projbuf, float* ffh, hipStream_t stream)
{
  const int M = T * N;
  for (int l = 0; l < L; ++l) {
    // qkv = x @ qkv_w[l] + qkv_b[l]
    k_gemm<<<dim3((3 * C_D) / 64, M / 128), 256, 0, stream>>>(
        x, p.qkv_w + (size_t)l * C_D * 3 * C_D, p.qkv_b + (size_t)l * 3 * C_D,
        qkvbuf, M, 3 * C_D, C_D, 0, 0, 0);
    // attention
    k_attn<<<dim3(N * C_HEADS), 128, 0, stream>>>(qkvbuf, attnctx, validk, T, N);
    // out projection
    k_gemm<<<dim3(C_D / 64, M / 128), 256, 0, stream>>>(
        attnctx, p.out_w + (size_t)l * C_D * C_D, p.out_b + (size_t)l * C_D,
        projbuf, M, C_D, C_D, 0, 0, 0);
    k_ln_res<<<M, 256, 0, stream>>>(x, projbuf, p.ln1_s + (size_t)l * C_D,
                                    p.ln1_b + (size_t)l * C_D);
    // FFN (chunked over rows to bound workspace)
    for (int off = 0; off < M; off += 8192) {
      int mc = imin_(8192, M - off);
      k_gemm<<<dim3(C_FFN / 64, mc / 128), 256, 0, stream>>>(
          x + (size_t)off * C_D, p.ff1_w + (size_t)l * C_D * C_FFN,
          p.ff1_b + (size_t)l * C_FFN, ffh, mc, C_FFN, C_D, 0, 1, 0);
      k_gemm<<<dim3(C_D / 64, mc / 128), 256, 0, stream>>>(
          ffh, p.ff2_w + (size_t)l * C_FFN * C_D, p.ff2_b + (size_t)l * C_D,
          projbuf + (size_t)off * C_D, mc, C_D, C_FFN, 0, 0, 0);
    }
    k_ln_res<<<M, 256, 0, stream>>>(x, projbuf, p.ln2_s + (size_t)l * C_D,
                                    p.ln2_b + (size_t)l * C_D);
  }
}

extern "C" void kernel_launch(void* const* d_in, const int* in_sizes, int n_in,
                              void* d_out, int out_size, void* d_ws, size_t ws_size,
                              hipStream_t stream)
{
  (void)in_sizes; (void)n_in; (void)ws_size;

  const int*   units    = (const int*)d_in[0];
  const int*   paths    = (const int*)d_in[1];
  const float* emb      = (const float*)d_in[2];
  const float* wih_f    = (const float*)d_in[3];
  const float* whh_f    = (const float*)d_in[4];
  const float* b_f      = (const float*)d_in[5];
  const float* wih_b    = (const float*)d_in[6];
  const float* whh_b    = (const float*)d_in[7];
  const float* b_b      = (const float*)d_in[8];
  const float* concat_w = (const float*)d_in[9];
  const float* attn_w   = (const float*)d_in[10];
  const float* attn_b   = (const float*)d_in[11];
  EncParams ep = make_enc(d_in, 12);
  EncParams fe = make_enc(d_in, 24);

  // ----- workspace carve -----
  char* wp = (char*)d_ws;
  auto allocf = [&](size_t n) { float* p = (float*)wp; wp += n * sizeof(float); return p; };

  float* x_emb   = allocf((size_t)C_TTOK * C_NU * C_D);     // 25.2M
  int*   ulen    = (int*)allocf(C_NU);
  float* h_f     = allocf((size_t)C_NU * C_D);
  float* c_f     = allocf((size_t)C_NU * C_D);
  float* h_b     = allocf((size_t)C_NU * C_D);
  float* c_b     = allocf((size_t)C_NU * C_D);
  float* G       = allocf((size_t)C_NU * C_4H);             // 4.2M
  float* Xstep   = allocf((size_t)C_NU * C_D);
  float* hcat    = allocf((size_t)C_NU * 2 * C_D);
  float* u       = allocf((size_t)C_NU * C_D);
  int*   endp    = (int*)allocf(C_NP);
  int*   contrib = (int*)allocf((size_t)C_TPATH * C_NP);
  int*   segarr  = (int*)allocf((size_t)C_TPATH * C_NP);
  int*   validk  = (int*)allocf((size_t)C_TPATH * C_NP);
  float* ctx     = allocf((size_t)C_TPATH * C_NP * C_D);    // 12.6M
  float* qkvbuf  = allocf((size_t)C_TPATH * C_NP * 3 * C_D);// 37.7M
  float* attnctx = allocf((size_t)C_TPATH * C_NP * C_D);    // 12.6M
  float* projbuf = allocf((size_t)C_TPATH * C_NP * C_D);    // 12.6M
  float* ffh     = allocf((size_t)8192 * C_FFN);            // 16.8M
  float* order_x = allocf((size_t)C_U * C_B * C_D);
  float* s_path  = allocf((size_t)C_TPATH * C_NP);
  float* s_old   = allocf((size_t)C_B * C_U);

  // ----- unit embedding + lengths -----
  k_ulen<<<(C_NU + 255) / 256, 256, 0, stream>>>(units, ulen);
  {
    int tot = C_TTOK * C_NU * C_D;
    k_embed<<<tot / 256, 256, 0, stream>>>(units, emb, x_emb);
  }

  // ----- BiLSTM (forward + backward over reversed valid sequence) -----
  const int NH = C_NU * C_D;
  auto lstm_dir = [&](const float* wih, const float* whh, const float* bb,
                      float* h, float* c, bool backward) {
    k_fill<<<NH / 256, 256, 0, stream>>>(h, 0.f, NH);
    k_fill<<<NH / 256, 256, 0, stream>>>(c, 0.f, NH);
    for (int t = 0; t < C_TTOK; ++t) {
      const float* Xt = x_emb + (size_t)t * C_NU * C_D;
      if (backward) {
        k_rev_step<<<NH / 256, 256, 0, stream>>>(Xstep, x_emb, ulen, t);
        Xt = Xstep;
      }
      // G = Xt @ wih^T + bb
      k_gemm<<<dim3(C_4H / 64, C_NU / 128), 256, 0, stream>>>(
          Xt, wih, bb, G, C_NU, C_4H, C_D, 1, 0, 0);
      // G += h @ whh^T
      k_gemm<<<dim3(C_4H / 64, C_NU / 128), 256, 0, stream>>>(
          h, whh, nullptr, G, C_NU, C_4H, C_D, 1, 0, 1);
      k_lstm_gate<<<NH / 256, 256, 0, stream>>>(G, h, c, ulen, t);
    }
  };
  lstm_dir(wih_f, whh_f, b_f, h_f, c_f, false);
  lstm_dir(wih_b, whh_b, b_b, h_b, c_b, true);

  // ----- u = [h_f, h_b] @ concat_w -----
  k_hcat<<<(C_NU * 2 * C_D) / 256, 256, 0, stream>>>(h_f, h_b, hcat);
  k_gemm<<<dim3(C_D / 64, C_NU / 128), 256, 0, stream>>>(
      hcat, concat_w, nullptr, u, C_NU, C_D, 2 * C_D, 0, 0, 0);

  // ----- path gather + posenc; path encoder (expaths, in-place in ctx) -----
  k_plen<<<(C_NP + 255) / 256, 256, 0, stream>>>(paths, endp);
  k_paths_meta<<<(C_TPATH * C_NP + 255) / 256, 256, 0, stream>>>(
      paths, endp, contrib, segarr, validk);
  k_build_ctx<<<(C_TPATH * C_NP * C_D) / 256, 256, 0, stream>>>(u, contrib, segarr, ctx);
  run_encoder(ctx, C_TPATH, C_NP, ep, 2, validk, qkvbuf, attnctx, projbuf, ffh, stream);

  // ----- order encoder (units per graph, no mask) -----
  k_build_order<<<(C_U * C_B * C_D) / 256, 256, 0, stream>>>(u, order_x);
  run_encoder(order_x, C_U, C_B, ep, 2, nullptr, qkvbuf, attnctx, projbuf, ffh, stream);

  // ----- splice: masked-softmax merge per (graph, unit) -----
  k_score<<<(C_TPATH * C_NP) / 8, 256, 0, stream>>>(
      ctx, attn_w, attn_b, s_path, C_TPATH * C_NP, 0);
  k_score<<<(C_B * C_U) / 8, 256, 0, stream>>>(
      order_x, attn_w, attn_b, s_old, C_B * C_U, 1);

  float* out = (float*)d_out;
  k_splice<<<C_B * C_U, 256, 0, stream>>>(ctx, order_x, s_path, s_old,
                                          contrib, segarr, out);

  // ----- final encoder, in place on d_out [U, B, D] -----
  run_encoder(out, C_U, C_B, fe, 2, nullptr, qkvbuf, attnctx, projbuf, ffh, stream);

  // ----- zero tail (final_mask region, all False) -----
  if (out_size > C_OUTEL) {
    int tail = out_size - C_OUTEL;
    k_fill<<<(tail + 255) / 256, 256, 0, stream>>>(out + C_OUTEL, 0.f, tail);
  }
}
